// MultipleCodebookVectorQuantizer_38319698215617
// MI455X (gfx1250) — matrix operand order repaired and checked by reading, not compiled, not run on hardware
//
#include <hip/hip_runtime.h>
#include <hip/hip_bf16.h>

typedef __bf16 bf16_t;
typedef __attribute__((ext_vector_type(16))) __bf16 v16bf;
typedef __attribute__((ext_vector_type(8)))  __bf16 v8bf;
typedef __attribute__((ext_vector_type(8)))  float  v8f;
typedef __attribute__((ext_vector_type(4)))  unsigned int u32x4;
typedef __attribute__((ext_vector_type(8)))  unsigned int u32x8;

#define BN   32
#define DCH  512
#define HW   1024      // 32*32
#define CC   4
#define KK   2048
#define DR   128
#define NTOK (BN * HW) // 32768
#define TN   256       // token tile (32 tokens per wave, 8 waves)
#define TK   128       // embedding tile
#define LDST 136       // padded LDS row stride in halfs (272 B: 256 B data + 16 B pad)

// ---------- prep: embeddings f32 -> bf16 workspace + row squared norms ----------
__global__ __launch_bounds__(128)
void vq_prep_kernel(const float* __restrict__ emb,
                    bf16_t* __restrict__ emb_bf,
                    float* __restrict__ e2) {
  const int row = blockIdx.x;          // c*KK + k  (8192 rows)
  const int d   = threadIdx.x;         // 0..127
  const float v = emb[(size_t)row * DR + d];
  emb_bf[(size_t)row * DR + d] = (bf16_t)v;
  float sq = v * v;
  #pragma unroll
  for (int m = 16; m >= 1; m >>= 1) sq += __shfl_xor(sq, m, 32);
  __shared__ float red[4];
  if ((threadIdx.x & 31) == 0) red[threadIdx.x >> 5] = sq;
  __syncthreads();
  if (threadIdx.x == 0) e2[row] = red[0] + red[1] + red[2] + red[3];
}

__global__ void vq_init_loss_kernel(float* loss) {
  if (threadIdx.x == 0) *loss = 0.0f;
}

// ---------- main: TDM-fed WMMA distance search + argmin + gather + loss ----------
__global__ __launch_bounds__(256)
void vq_main_kernel(const float* __restrict__ latents,
                    const float* __restrict__ emb_f32,
                    const bf16_t* __restrict__ emb_bf,
                    const float* __restrict__ e2,
                    float* __restrict__ out,
                    float* __restrict__ loss) {
  __shared__ __align__(256) bf16_t Ash[TN * LDST];
  __shared__ __align__(256) bf16_t Bsh[TK * LDST];
  __shared__ float  e2sh[TK];
  __shared__ int    idxsh[TN];
  __shared__ float  redsh[8];

  const int bid  = blockIdx.x;
  const int c    = bid >> 7;              // codebook 0..3
  const int tile = bid & 127;             // token tile 0..127
  const int n0   = tile * TN;
  const int bimg = n0 >> 10;              // image index (TN=256 divides HW=1024)
  const int hw0  = n0 & 1023;
  const int t    = threadIdx.x;
  const size_t latBase = (size_t)bimg * (DCH * HW) + hw0;

  // Load A tile: flat[c, n0+tok, d] = latents[bimg, c+d, hw0+tok], f32 -> bf16
  for (int i = t; i < TN * DR; i += 256) {
    const int d = i >> 8, tok = i & 255;
    const float v = latents[latBase + (size_t)(c + d) * HW + tok];
    Ash[tok * LDST + d] = (bf16_t)v;
  }
  __syncthreads();

  const int wave = t >> 5;
  const int lane = t & 31;
  const int lm   = lane & 15;
  const int sel  = lane >> 4;

  // Preload 2 sets x 4 A-fragments (32 tokens per wave, Dr=128 = 4 chunks of K=32).
  // Dense 16-bit A 16x32 layout: lanes<16 -> [K0..K7, K16..K23]; lanes>=16 -> [K8..K15, K24..K31]
  v16bf afrag[2][4];
  #pragma unroll
  for (int s = 0; s < 2; ++s) {
    const int arow = wave * 32 + s * 16 + lm;
    #pragma unroll
    for (int kc = 0; kc < 4; ++kc) {
      const bf16_t* p = &Ash[arow * LDST + kc * 32];
      const v8bf lo = *reinterpret_cast<const v8bf*>(p + 8 * sel);
      const v8bf hi = *reinterpret_cast<const v8bf*>(p + 16 + 8 * sel);
      #pragma unroll
      for (int i = 0; i < 8; ++i) { afrag[s][kc][i] = lo[i]; afrag[s][kc][8 + i] = hi[i]; }
    }
  }

  float minD[2][8];
  int   minI[2][8];
  #pragma unroll
  for (int s = 0; s < 2; ++s)
    #pragma unroll
    for (int v = 0; v < 8; ++v) { minD[s][v] = 3.4e38f; minI[s][v] = 0; }

  // Static parts of the TDM descriptor (D#): 2D tensor of bf16, 128x128 tile,
  // row = 128 elems (256 B), LDS pad of 4 DWORDs (16 B) every 64 DWORDs (256 B)
  // -> matches LDST=136-half padded rows.
  const unsigned ldsAddr = (unsigned)(unsigned long long)(const void*)Bsh;
  u32x8 g1;
  g1[0] = (1u << 16)          // data_size = 2 bytes
        | (1u << 20)          // pad_enable
        | (5u << 22)          // pad_interval: 8*2^5 = 256 B
        | (3u << 25);         // pad_amount: 4 DWORDs = 16 B
  g1[1] = (unsigned)(DR) << 16;            // tensor_dim0[15:0] = 128
  g1[2] = ((unsigned)KK & 0xFFFFu) << 16;  // tensor_dim0[31:16]=0 | tensor_dim1[15:0]=2048
  g1[3] = (unsigned)TK << 16;              // tensor_dim1[31:16]=0 | tile_dim0 = 128
  g1[4] = (unsigned)TK;                    // tile_dim1 = 128, tile_dim2 = 0
  g1[5] = (unsigned)DR;                    // tensor_dim0_stride[31:0] = 128 elems
  g1[6] = 0u;                              // stride0 hi | tensor_dim1_stride lo
  g1[7] = 0u;                              // tensor_dim1_stride hi

  for (int kt = 0; kt < KK / TK; ++kt) {
    __syncthreads();  // previous B tile fully consumed
    if (wave == 0) {
      // TDM: async-load the 32 KB B tile (with LDS row padding) from L2-resident ws
      const unsigned long long ga =
          (unsigned long long)(const void*)(emb_bf + ((size_t)(c * KK + kt * TK)) * DR);
      u32x4 g0;
      g0[0] = 1u;                      // count = 1 valid descriptor
      g0[1] = ldsAddr;                 // lds_addr
      g0[2] = (unsigned)(ga & 0xFFFFFFFFull);
      g0[3] = (unsigned)((ga >> 32) & 0x1FFFFFFull) | (2u << 30);  // addr[56:32] | type=2
      asm volatile("tensor_load_to_lds %0, %1" :: "s"(g0), "s"(g1) : "memory");
      __builtin_amdgcn_s_wait_tensorcnt(0);
    }
    if (t < TK) e2sh[t] = e2[c * KK + kt * TK + t];
    __syncthreads();

    #pragma unroll
    for (int ec = 0; ec < 8; ++ec) {          // 8 chunks of 16 embeddings
      v8f acc0 = {};
      v8f acc1 = {};
      #pragma unroll
      for (int kc = 0; kc < 4; ++kc) {
        // Dense 16-bit B 32x16 layout: lanes<16 -> [K0..K7, K16..K23]; lanes>=16 -> [K8..K15, K24..K31], N = lane%16
        const bf16_t* p = &Bsh[(ec * 16 + lm) * LDST + kc * 32 + 8 * sel];
        const v8bf lo = *reinterpret_cast<const v8bf*>(p);
        const v8bf hi = *reinterpret_cast<const v8bf*>(p + 16);
        v16bf bfrag;
        #pragma unroll
        for (int i = 0; i < 8; ++i) { bfrag[i] = lo[i]; bfrag[8 + i] = hi[i]; }
        acc0 = __builtin_amdgcn_wmma_f32_16x16x32_bf16(
            false, afrag[0][kc], false, bfrag, (short)0, acc0, false, false);
        acc1 = __builtin_amdgcn_wmma_f32_16x16x32_bf16(
            false, afrag[1][kc], false, bfrag, (short)0, acc1, false, false);
      }
      const int   eidx  = kt * TK + ec * 16 + lm;
      const float ecost = e2sh[ec * 16 + lm];
      #pragma unroll
      for (int v = 0; v < 8; ++v) {
        const float d0 = ecost - 2.0f * acc0[v];   // x^2 constant per token: drop it
        if (d0 < minD[0][v]) { minD[0][v] = d0; minI[0][v] = eidx; }
        const float d1 = ecost - 2.0f * acc1[v];
        if (d1 < minD[1][v]) { minD[1][v] = d1; minI[1][v] = eidx; }
      }
    }
  }

  // Cross-lane argmin within each 16-lane group (lane holds embedding column lane%16;
  // D layout: VGPR v -> token row v (lanes 0-15) / v+8 (lanes 16-31)).
  #pragma unroll
  for (int s = 0; s < 2; ++s) {
    #pragma unroll
    for (int v = 0; v < 8; ++v) {
      #pragma unroll
      for (int m = 8; m >= 1; m >>= 1) {
        const float od = __shfl_xor(minD[s][v], m, 32);
        const int   oi = __shfl_xor(minI[s][v], m, 32);
        if (od < minD[s][v] || (od == minD[s][v] && oi < minI[s][v])) {
          minD[s][v] = od; minI[s][v] = oi;
        }
      }
    }
    if (lm == 0) {
      #pragma unroll
      for (int v = 0; v < 8; ++v) idxsh[wave * 32 + s * 16 + sel * 8 + v] = minI[s][v];
    }
  }
  __syncthreads();

  // Gather winning f32 rows, write output [B, C*DR, H, W], accumulate loss
  float lsum = 0.0f;
  for (int i = t; i < TN * DR; i += 256) {
    const int d = i >> 8, tok = i & 255;
    const int e = idxsh[tok];
    const float q = emb_f32[((size_t)(c * KK + e)) * DR + d];
    const float x = latents[latBase + (size_t)(c + d) * HW + tok];
    out[latBase + (size_t)(c * DR + d) * HW + tok] = q;
    const float df = q - x;
    lsum += df * df;
  }
  #pragma unroll
  for (int m = 16; m >= 1; m >>= 1) lsum += __shfl_xor(lsum, m, 32);
  if (lane == 0) redsh[wave] = lsum;
  __syncthreads();
  if (t == 0) {
    float s = 0.0f;
    #pragma unroll
    for (int w = 0; w < 8; ++w) s += redsh[w];
    // vq_loss = sum_c (beta+1) * mean((q-flat)^2);  mean over N*DR per codebook
    atomicAdd(loss, s * (1.25f / ((float)NTOK * (float)DR)));
  }
}

extern "C" void kernel_launch(void* const* d_in, const int* in_sizes, int n_in,
                              void* d_out, int out_size, void* d_ws, size_t ws_size,
                              hipStream_t stream) {
  const float* latents = (const float*)d_in[0];      // [32, 512, 32, 32]
  const float* emb     = (const float*)d_in[1];      // [4, 2048, 128]
  float* out  = (float*)d_out;                       // quantized [32,512,32,32] then loss scalar
  float* loss = out + (size_t)BN * DCH * HW;

  bf16_t* emb_bf = (bf16_t*)d_ws;                                    // 2 MB
  float*  e2     = (float*)((char*)d_ws + (size_t)CC * KK * DR * 2); // 32 KB

  vq_init_loss_kernel<<<1, 32, 0, stream>>>(loss);
  vq_prep_kernel<<<CC * KK, 128, 0, stream>>>(emb, emb_bf, e2);
  vq_main_kernel<<<CC * (NTOK / TN), 256, 0, stream>>>(latents, emb, emb_bf, e2, out, loss);
}
